// EquivariantThreeHopGINE_7112465842229
// MI455X (gfx1250) — matrix operand order repaired and checked by reading, not compiled, not run on hardware
//
#include <hip/hip_runtime.h>

// EquivariantThreeHopGINE forward for MI455X (gfx1250, wave32, WMMA).
// GEMMs in f16 WMMA with f32 accumulation (memory-bound workload: precision
// downgrade on the matmul inputs is free perf; graph scatter stays fp32 via
// hardware global_atomic_add_f32).

typedef __attribute__((ext_vector_type(16))) _Float16 v16h;
typedef __attribute__((ext_vector_type(8)))  _Float16 v8h;
typedef __attribute__((ext_vector_type(4)))  _Float16 v4h;
typedef __attribute__((ext_vector_type(8)))  float    v8f;

#define HDIM 256

// ---------------------------------------------------------------- fragments
// A-matrix 16x32 f16 (ISA 7.12.2): lane L: row = L&15, g = L>>4;
//   halves j=0..7  -> K = k0 + 8*g + j
//   halves j=8..15 -> K = k0 + 16 + 8*g + (j-8)
// B-matrix 32x16 f16: lane L: col = L&15, g = L>>4;
//   halves j=0..7 -> K = k0 + 8*g + j; j=8..15 -> K = k0 + 16 + 8*g + (j-8)
__device__ __forceinline__ v16h ld_frag(const _Float16* p0, const _Float16* p1) {
  v8h lo = *(const v8h*)p0;
  v8h hi = *(const v8h*)p1;
  v16h r;
#pragma unroll
  for (int i = 0; i < 8; ++i) { r[i] = lo[i]; r[i + 8] = hi[i]; }
  return r;
}

// ------------------------------------------------------------------- GEMM
// C[M,256] = A[M,K] (f16, row-major) x B[K,256] + bias, f32 out.
// BT is pre-transposed: BT[n*K + k] = B[k][n] so all fragment loads are
// contiguous 16B. Wave -> 16x64 C tile; block (8 waves) -> 128x64.
template <int K>
__global__ __launch_bounds__(256) void gine_gemm_wmma(
    const _Float16* __restrict__ A,
    const _Float16* __restrict__ BT,
    const float* __restrict__ bias,
    float* __restrict__ C,
    int M)
{
  const int lane = threadIdx.x & 31;
  const int wave = threadIdx.x >> 5;
  const int g    = lane >> 4;
  const int l16  = lane & 15;

  const int mbase = blockIdx.x * 128 + wave * 16;
  const int n0    = blockIdx.y * 64;

  int arow = mbase + l16; if (arow >= M) arow = M - 1;   // clamp, store-guarded
  const _Float16* Ap = A + (size_t)arow * K + 8 * g;

  const _Float16* Bp[4];
#pragma unroll
  for (int nf = 0; nf < 4; ++nf)
    Bp[nf] = BT + (size_t)(n0 + nf * 16 + l16) * K + 8 * g;

  v8f acc[4] = {v8f{}, v8f{}, v8f{}, v8f{}};

#pragma unroll
  for (int k0 = 0; k0 < K; k0 += 32) {
    if (k0 + 32 < K) __builtin_prefetch(Ap + k0 + 32, 0, 1);  // global_prefetch
    v16h a = ld_frag(Ap + k0, Ap + k0 + 16);
#pragma unroll
    for (int nf = 0; nf < 4; ++nf) {
      v16h b = ld_frag(Bp[nf] + k0, Bp[nf] + k0 + 16);
      acc[nf] = __builtin_amdgcn_wmma_f32_16x16x32_f16(
          false, a, false, b, (short)0, acc[nf], false, false);
    }
  }

  // C/D layout: VGPR r: lanes 0-15 -> M = r, lanes 16-31 -> M = r + 8.
#pragma unroll
  for (int nf = 0; nf < 4; ++nf) {
    const int col = n0 + nf * 16 + l16;
    const float bv = bias[col];
#pragma unroll
    for (int r = 0; r < 8; ++r) {
      const int row = mbase + g * 8 + r;
      if (row < M) C[(size_t)row * HDIM + col] = acc[nf][r] + bv;
    }
  }
}

// ------------------------------------------------------- weight transpose
// WT[n*Kpad + k] = (k < Krows) ? W[k*256 + n] : 0   (f32 -> f16, once/launch)
__global__ void gine_transpose_cvt(const float* __restrict__ W,
                                   _Float16* __restrict__ WT,
                                   int Krows, int Kpad)
{
  int idx = blockIdx.x * blockDim.x + threadIdx.x;
  if (idx >= HDIM * Kpad) return;
  int n = idx / Kpad, k = idx - n * Kpad;
  WT[(size_t)n * Kpad + k] =
      (k < Krows) ? (_Float16)W[(size_t)k * HDIM + n] : (_Float16)0.0f;
}

// --------------------------------------------------------- atom embedding
// feats16[N,96]: concat of 7 table lookups (88 cols) + 8 zero-pad cols.
__global__ void gine_embed(const int* __restrict__ ai,
                           const float* __restrict__ e_el,
                           const float* __restrict__ e_de,
                           const float* __restrict__ e_va,
                           const float* __restrict__ e_ch,
                           const float* __restrict__ e_ar,
                           const float* __restrict__ e_hy,
                           const float* __restrict__ e_hh,
                           _Float16* __restrict__ feats, int N)
{
  int idx = blockIdx.x * blockDim.x + threadIdx.x;
  int node = idx >> 7;            // 128 slots per node, 96 used
  int j    = idx & 127;
  if (node >= N || j >= 96) return;
  const int* a = ai + (size_t)node * 7;
  float v = 0.0f;
  if      (j < 64) v = e_el[a[0] * 64 + j];
  else if (j < 68) v = e_de[a[1] * 4 + (j - 64)];
  else if (j < 72) v = e_va[(a[2] + 1) * 4 + (j - 68)];
  else if (j < 76) v = e_ch[a[3] * 4 + (j - 72)];
  else if (j < 80) v = e_ar[a[4] * 4 + (j - 76)];
  else if (j < 84) v = e_hy[a[5] * 4 + (j - 80)];
  else if (j < 88) v = e_hh[a[6] * 4 + (j - 84)];
  feats[(size_t)node * 96 + j] = (_Float16)v;
}

// ------------------------------------------- edge gather + relu + scatter
// msg = relu(h[s] + (We+be)); agg[d] += msg. 64 lanes (float4) per edge,
// 4 edges per 256-thread block. fp32 hardware atomics (order-nondeterministic
// summation, standard for GNN scatter).
__global__ __launch_bounds__(256) void gine_edge_scatter(
    const float* __restrict__ h,
    const int* __restrict__ src, const int* __restrict__ dst,
    const float* __restrict__ We, const float* __restrict__ be,
    float* __restrict__ agg, int E)
{
  int tid  = threadIdx.x;
  int e    = blockIdx.x * 4 + (tid >> 6);
  int lane = tid & 63;
  int col  = lane * 4;
  if (e >= 2 * E) return;
  int s_, d_;
  if (e < E) { s_ = src[e]; d_ = dst[e]; }
  else       { s_ = dst[e - E]; d_ = src[e - E]; }

  const float4 x = *(const float4*)(h + (size_t)s_ * HDIM + col);
  float m0 = fmaxf(x.x + We[col + 0] + be[col + 0], 0.0f);
  float m1 = fmaxf(x.y + We[col + 1] + be[col + 1], 0.0f);
  float m2 = fmaxf(x.z + We[col + 2] + be[col + 2], 0.0f);
  float m3 = fmaxf(x.w + We[col + 3] + be[col + 3], 0.0f);
  float* ap = agg + (size_t)d_ * HDIM + col;
  unsafeAtomicAdd(ap + 0, m0);   // global_atomic_add_f32 (no return)
  unsafeAtomicAdd(ap + 1, m1);
  unsafeAtomicAdd(ap + 2, m2);
  unsafeAtomicAdd(ap + 3, m3);
}

// ----------------------------------------------------- x16 = f16(h + agg)
__global__ void gine_addcvt(const float* __restrict__ h,
                            const float* __restrict__ agg,
                            _Float16* __restrict__ x16, long n4)
{
  long i = (long)blockIdx.x * blockDim.x + threadIdx.x;
  if (i >= n4) return;
  float4 a = ((const float4*)h)[i];
  float4 b = ((const float4*)agg)[i];
  v4h o;
  o[0] = (_Float16)(a.x + b.x);
  o[1] = (_Float16)(a.y + b.y);
  o[2] = (_Float16)(a.z + b.z);
  o[3] = (_Float16)(a.w + b.w);
  ((v4h*)x16)[i] = o;
}

__global__ void gine_cvt(const float* __restrict__ h,
                         _Float16* __restrict__ x16, long n4)
{
  long i = (long)blockIdx.x * blockDim.x + threadIdx.x;
  if (i >= n4) return;
  float4 a = ((const float4*)h)[i];
  v4h o;
  o[0] = (_Float16)a.x; o[1] = (_Float16)a.y;
  o[2] = (_Float16)a.z; o[3] = (_Float16)a.w;
  ((v4h*)x16)[i] = o;
}

// --------------------------------------------------------------- LayerNorm
// One node per block (256 threads = 8 wave32). Wave shuffle + LDS reduction.
__global__ __launch_bounds__(256) void gine_layernorm(
    const float* __restrict__ y,
    const float* __restrict__ gam, const float* __restrict__ bet,
    float* __restrict__ h, int N)
{
  __shared__ float ssum[8], ssq[8];
  int node = blockIdx.x;
  if (node >= N) return;
  int tid = threadIdx.x;
  float v = y[(size_t)node * HDIM + tid];
  float s = v, q = v * v;
#pragma unroll
  for (int m = 16; m >= 1; m >>= 1) {
    s += __shfl_xor(s, m, 32);
    q += __shfl_xor(q, m, 32);
  }
  int wave = tid >> 5, lane = tid & 31;
  if (lane == 0) { ssum[wave] = s; ssq[wave] = q; }
  __syncthreads();
  if (wave == 0) {
    float s2 = (lane < 8) ? ssum[lane] : 0.0f;
    float q2 = (lane < 8) ? ssq[lane] : 0.0f;
#pragma unroll
    for (int m = 4; m >= 1; m >>= 1) {
      s2 += __shfl_xor(s2, m, 32);
      q2 += __shfl_xor(q2, m, 32);
    }
    if (lane == 0) { ssum[0] = s2; ssq[0] = q2; }
  }
  __syncthreads();
  float mean = ssum[0] * (1.0f / HDIM);
  float var  = ssq[0] * (1.0f / HDIM) - mean * mean;
  float rstd = rsqrtf(var + 1e-5f);
  h[(size_t)node * HDIM + tid] = (v - mean) * rstd * gam[tid] + bet[tid];
}

// ------------------------------------------------------------------ launch
extern "C" void kernel_launch(void* const* d_in, const int* in_sizes, int n_in,
                              void* d_out, int out_size, void* d_ws, size_t ws_size,
                              hipStream_t stream)
{
  const int N = in_sizes[0] / 7;
  const int E = in_sizes[1];

  const int*   atom = (const int*)d_in[0];
  const int*   src  = (const int*)d_in[1];
  const int*   dst  = (const int*)d_in[2];
  const float* e_el = (const float*)d_in[3];
  const float* e_de = (const float*)d_in[4];
  const float* e_va = (const float*)d_in[5];
  const float* e_ch = (const float*)d_in[6];
  const float* e_ar = (const float*)d_in[7];
  const float* e_hy = (const float*)d_in[8];
  const float* e_hh = (const float*)d_in[9];
  const float* W0   = (const float*)d_in[10];
  const float* b0   = (const float*)d_in[11];
  const float* W1   = (const float*)d_in[12];
  const float* b1   = (const float*)d_in[13];
  const float *Wn[4], *bn[4], *We[4], *be[4], *gam[4], *bet[4];
  for (int i = 0; i < 4; ++i) {
    Wn[i]  = (const float*)d_in[14 + 6 * i];
    bn[i]  = (const float*)d_in[15 + 6 * i];
    We[i]  = (const float*)d_in[16 + 6 * i];
    be[i]  = (const float*)d_in[17 + 6 * i];
    gam[i] = (const float*)d_in[18 + 6 * i];
    bet[i] = (const float*)d_in[19 + 6 * i];
  }

  // ---- workspace layout (all offsets 256B-aligned)
  const size_t NH = (size_t)N * HDIM;
  char* p = (char*)d_ws;
  float*    h32 = (float*)p;              p += NH * sizeof(float);
  float*    agg = (float*)p;              p += NH * sizeof(float);
  _Float16* x16 = (_Float16*)p;           p += NH * sizeof(_Float16);
  _Float16* W0T = (_Float16*)p;           p += (size_t)HDIM * 96 * sizeof(_Float16);
  _Float16* WT  = (_Float16*)p;           // 5 x 256*256 f16 slots
  _Float16* feats = (_Float16*)agg;       // alias: only live before 1st memset

  // ---- 1) weight transpose+convert (tiny, once per launch)
  {
    int t0 = HDIM * 96;
    gine_transpose_cvt<<<(t0 + 255) / 256, 256, 0, stream>>>(W0, W0T, 88, 96);
    int t1 = HDIM * HDIM;
    for (int i = 0; i < 4; ++i)
      gine_transpose_cvt<<<(t1 + 255) / 256, 256, 0, stream>>>(
          Wn[i], WT + (size_t)i * HDIM * HDIM, HDIM, HDIM);
    gine_transpose_cvt<<<(t1 + 255) / 256, 256, 0, stream>>>(
        W1, WT + (size_t)4 * HDIM * HDIM, HDIM, HDIM);
  }

  // ---- 2) embedding concat -> f16 feats[N,96]
  {
    long total = (long)N * 128;
    gine_embed<<<(unsigned)((total + 255) / 256), 256, 0, stream>>>(
        atom, e_el, e_de, e_va, e_ch, e_ar, e_hy, e_hh, feats, N);
  }

  // ---- 3) h = feats @ W0 + b0   (WMMA, K=96)
  dim3 ggrid((N + 127) / 128, 4);
  gine_gemm_wmma<96><<<ggrid, 256, 0, stream>>>(feats, W0T, b0, h32, N);

  // ---- 4) four GINE layers
  const long n4 = (long)NH / 4;
  const unsigned cvtBlocks  = (unsigned)((n4 + 255) / 256);
  const unsigned edgeBlocks = (unsigned)((2L * E + 3) / 4);
  for (int i = 0; i < 4; ++i) {
    hipMemsetAsync(agg, 0, NH * sizeof(float), stream);
    gine_edge_scatter<<<edgeBlocks, 256, 0, stream>>>(
        h32, src, dst, We[i], be[i], agg, E);
    gine_addcvt<<<cvtBlocks, 256, 0, stream>>>(h32, agg, x16, n4);
    // y = (h + agg) @ Wn + bn   -> reuse agg as y
    gine_gemm_wmma<HDIM><<<ggrid, 256, 0, stream>>>(
        x16, WT + (size_t)i * HDIM * HDIM, bn[i], agg, N);
    gine_layernorm<<<N, 256, 0, stream>>>(agg, gam[i], bet[i], h32, N);
  }

  // ---- 5) out = h @ W1 + b1
  gine_cvt<<<cvtBlocks, 256, 0, stream>>>(h32, x16, n4);
  gine_gemm_wmma<HDIM><<<ggrid, 256, 0, stream>>>(
      x16, WT + (size_t)4 * HDIM * HDIM, b1, (float*)d_out, N);
}
